// Hydra_30674656428539
// MI455X (gfx1250) — matrix operand order repaired and checked by reading, not compiled, not run on hardware
//
#include <hip/hip_runtime.h>
#include <hip/hip_bf16.h>

typedef __attribute__((ext_vector_type(16))) _Float16 v16h;
typedef __attribute__((ext_vector_type(8)))  _Float16 v8h;
typedef __attribute__((ext_vector_type(8)))  float    v8f;

// ---------------- model constants ----------------
#define LSEQ      2048
#define DMODEL    1024
#define DINNER    2048
#define NHEADS    32
#define HEADDIM   64
#define DSTATE    64
#define CHUNKSZ   256
#define NCHUNK    8          // LSEQ / CHUNKSZ
#define BATCH     2
#define NBD       4          // 2*BATCH (fw+bw)
#define DINPROJ   4416
#define CONVDIM   2304
#define MROWS     (BATCH*LSEQ)   // 4096

// ---------------- workspace layout (floats) ----------------
static const size_t OFF_ZX  = 0;                            // zxbcdt   [4096][4416]
static const size_t OFF_DT  = OFF_ZX  + (size_t)MROWS*DINPROJ;          // dt2 [4][2048][32]
static const size_t OFF_XBC = OFF_DT  + (size_t)NBD*LSEQ*NHEADS;        // xBC post-conv [2][2048][2304]
static const size_t OFF_ACS = OFF_XBC + (size_t)BATCH*LSEQ*CONVDIM;     // Acs [4][32][8][256]
static const size_t OFF_CHS = OFF_ACS + (size_t)NBD*NHEADS*NCHUNK*CHUNKSZ; // chunk sums [4][32][8]
static const size_t OFF_YS  = OFF_CHS + (size_t)NBD*NHEADS*NCHUNK;      // y_ssd [4][2048][2048]
static const size_t OFF_ST  = OFF_YS  + (size_t)NBD*LSEQ*DINNER;        // states [4][32][8][64][64]
static const size_t OFF_SR  = OFF_ST  + (size_t)NBD*NHEADS*NCHUNK*HEADDIM*DSTATE; // S_run
static const size_t OFF_FCD = OFF_SR  + (size_t)NBD*NHEADS*NCHUNK*HEADDIM*DSTATE; // fcD [2][2048][32]
static const size_t OFF_YN  = OFF_FCD + (size_t)BATCH*LSEQ*NHEADS;      // normed gated y [2][2048][2048]

// ---------------- helpers ----------------
__device__ inline v8f zero8() {
  v8f z;
#pragma unroll
  for (int i = 0; i < 8; i++) z[i] = 0.0f;
  return z;
}

// A-frag (16x32 f16, MxK): lane m=lane&15, hh=lane>>4; halves: [hh*8,+8) and [16+hh*8,+8)
__device__ inline v16h afrag_f32(const float* __restrict__ row, int hh) {
  const float* p0 = row + hh * 8;
  const float* p1 = row + 16 + hh * 8;
  v16h a;
#pragma unroll
  for (int i = 0; i < 8; i++) { a[i] = (_Float16)p0[i]; a[8 + i] = (_Float16)p1[i]; }
  return a;
}
__device__ inline v16h afrag_h(const _Float16* __restrict__ row, int hh) {
  const _Float16* p0 = row + hh * 8;
  const _Float16* p1 = row + 16 + hh * 8;
  v16h a;
#pragma unroll
  for (int i = 0; i < 8; i++) { a[i] = p0[i]; a[8 + i] = p1[i]; }
  return a;
}
// B-frag (32x16 f16, KxN): lane n=lane&15, hh=lane>>4; halves: [hh*8,+8) and [16+hh*8,+8)
__device__ inline v16h bfrag_h(const _Float16* __restrict__ kcol, int hh) {
  const _Float16* p0 = kcol + hh * 8;
  const _Float16* p1 = kcol + 16 + hh * 8;
  v16h b;
#pragma unroll
  for (int i = 0; i < 8; i++) { b[i] = p0[i]; b[8 + i] = p1[i]; }
  return b;
}

#define WMMA_F16(A, B, C) \
  __builtin_amdgcn_wmma_f32_16x16x32_f16(false, (A), false, (B), (short)0, (C), false, false)

// ---------------- 1) GEMM: C[m,n] = sum_k A[m,k]*W[n,k] ----------------
// 256 threads (8 waves); block tile 128M x 64N; K-steps of 32, LDS double-buffered.
// Wave w computes M strip [w*16, w*16+16) x 64N (4 WMMA per K-step).
__global__ void __launch_bounds__(256) wmma_gemm_xwT(const float* __restrict__ A,
                                                     const float* __restrict__ W,
                                                     float* __restrict__ C,
                                                     int M, int N, int K) {
  __shared__ __align__(16) _Float16 sA[2][128 * 32];  // [row][k]
  __shared__ __align__(16) _Float16 sW[2][64 * 32];   // [n][k]
  const int tid = threadIdx.x, wave = tid >> 5, lane = tid & 31;
  const int col = lane & 15, hh = lane >> 4;
  const int mBlock = blockIdx.y * 128;
  const int nBlock = blockIdx.x * 64;

  // staging coords: thread loads 16 consecutive f32 of A, 8 consecutive f32 of W
  const int aRow = tid >> 1, aK = (tid & 1) * 16;
  const int wRow = tid >> 2, wK = (tid & 3) * 8;
  const float* aSrc = A + (size_t)(mBlock + aRow) * K + aK;
  const float* wSrc = W + (size_t)(nBlock + wRow) * K + wK;

  v8f acc[4];
#pragma unroll
  for (int i = 0; i < 4; i++) acc[i] = zero8();

  float aReg[16], wReg[8];
  // preload K-step 0
#pragma unroll
  for (int i = 0; i < 16; i++) aReg[i] = aSrc[i];
#pragma unroll
  for (int i = 0; i < 8; i++)  wReg[i] = wSrc[i];
  {
    _Float16* pa = &sA[0][aRow * 32 + aK];
#pragma unroll
    for (int i = 0; i < 16; i++) pa[i] = (_Float16)aReg[i];
    _Float16* pw = &sW[0][wRow * 32 + wK];
#pragma unroll
    for (int i = 0; i < 8; i++) pw[i] = (_Float16)wReg[i];
  }
  __syncthreads();

  const int nK = K >> 5;
  for (int k = 0; k < nK; k++) {
    const int buf = k & 1;
    if (k + 1 < nK) {
      const float* an = aSrc + (k + 1) * 32;
      const float* wn = wSrc + (k + 1) * 32;
      __builtin_prefetch(an + 32, 0, 1);
#pragma unroll
      for (int i = 0; i < 16; i++) aReg[i] = an[i];
#pragma unroll
      for (int i = 0; i < 8; i++)  wReg[i] = wn[i];
    }
    // compute from current buffer while next-tile loads are in flight
    v16h af = afrag_h(&sA[buf][(wave * 16 + col) * 32], hh);
#pragma unroll
    for (int nt = 0; nt < 4; nt++) {
      v16h bf = bfrag_h(&sW[buf][(nt * 16 + col) * 32], hh);
      acc[nt] = WMMA_F16(af, bf, acc[nt]);
    }
    __syncthreads();
    if (k + 1 < nK) {
      _Float16* pa = &sA[buf ^ 1][aRow * 32 + aK];
#pragma unroll
      for (int i = 0; i < 16; i++) pa[i] = (_Float16)aReg[i];
      _Float16* pw = &sW[buf ^ 1][wRow * 32 + wK];
#pragma unroll
      for (int i = 0; i < 8; i++) pw[i] = (_Float16)wReg[i];
    }
    __syncthreads();
  }

#pragma unroll
  for (int nt = 0; nt < 4; nt++) {
#pragma unroll
    for (int r = 0; r < 8; r++) {
      int row = mBlock + wave * 16 + r + 8 * hh;
      int cc  = nBlock + nt * 16 + col;
      C[(size_t)row * N + cc] = acc[nt][r];
    }
  }
}

// ---------------- 2) dt: split directions + softplus ----------------
__global__ void __launch_bounds__(256) prep_dt_kernel(const float* __restrict__ ZX,
                                                      const float* __restrict__ dt_bias,
                                                      float* __restrict__ DTo) {
  int idx = blockIdx.x * 256 + threadIdx.x;  // [4][2048][32]
  int hh = idx & 31;
  int l  = (idx >> 5) & (LSEQ - 1);
  int e  = idx >> 16;
  float raw;
  if (e < 2) raw = ZX[(size_t)(e * LSEQ + l) * DINPROJ + 4352 + hh];
  else       raw = ZX[(size_t)((e - 2) * LSEQ + (LSEQ - 1 - l)) * DINPROJ + 4384 + hh];
  float x = raw + dt_bias[hh];
  DTo[idx] = (x > 20.0f) ? x : log1pf(__expf(x));
}

// ---------------- 3) depthwise conv7 + silu ----------------
__global__ void __launch_bounds__(256) conv_silu_kernel(const float* __restrict__ ZX,
                                                        const float* __restrict__ conv_w,
                                                        const float* __restrict__ conv_b,
                                                        float* __restrict__ XBC) {
  int idx = blockIdx.x * 256 + threadIdx.x;  // [2][2048][2304]
  int c = idx % CONVDIM;
  int rem = idx / CONVDIM;
  int l = rem & (LSEQ - 1);
  int b = rem >> 11;
  float acc = conv_b[c];
#pragma unroll
  for (int j = 0; j < 7; j++) {
    int ll = l + j - 3;
    if (ll >= 0 && ll < LSEQ)
      acc += ZX[(size_t)(b * LSEQ + ll) * DINPROJ + DINNER + c] * conv_w[c * 7 + j];
  }
  XBC[idx] = acc / (1.0f + __expf(-acc));
}

// ---------------- 4) per-chunk cumsum of dt*A ----------------
__global__ void __launch_bounds__(256) chunk_cumsum_kernel(const float* __restrict__ DT,
                                                           const float* __restrict__ A_log,
                                                           float* __restrict__ ACS,
                                                           float* __restrict__ CHS) {
  int bid = blockIdx.x;  // e*256 + h*8 + c
  int e = bid >> 8, h = (bid >> 3) & 31, c = bid & 7;
  int t = threadIdx.x;
  __shared__ float s[CHUNKSZ];
  float A = -__expf(A_log[h]);
  s[t] = DT[(size_t)(e * LSEQ + c * CHUNKSZ + t) * NHEADS + h] * A;
  __syncthreads();
  for (int o = 1; o < CHUNKSZ; o <<= 1) {
    float add = (t >= o) ? s[t - o] : 0.0f;
    __syncthreads();
    s[t] += add;
    __syncthreads();
  }
  ACS[(size_t)((e * NHEADS + h) * NCHUNK + c) * CHUNKSZ + t] = s[t];
  if (t == 255) CHS[(e * NHEADS + h) * NCHUNK + c] = s[255];
}

// ---------------- 5) SSD intra-chunk: Y_diag + chunk states (WMMA) ----------------
__global__ void __launch_bounds__(256) ssd_chunk_kernel(const float* __restrict__ XBC,
                                                        const float* __restrict__ DT,
                                                        const float* __restrict__ ACS,
                                                        float* __restrict__ YS,
                                                        float* __restrict__ STATES) {
  int bid = blockIdx.x;
  int e = bid >> 8, h = (bid >> 3) & 31, c = bid & 7;
  int tid = threadIdx.x, wave = tid >> 5, lane = tid & 31;
  int col = lane & 15, hh = lane >> 4;

  __shared__ __align__(16) float    sAcs[CHUNKSZ];
  __shared__ __align__(16) _Float16 sB[32 * 64];    // [s_local][n]
  __shared__ __align__(16) _Float16 sBT[64 * 32];   // [n][s_local]
  __shared__ __align__(16) _Float16 sXT[64 * 32];   // [p][s_local]  (x*dt)
  __shared__ __align__(16) _Float16 sXdT[64 * 32];  // [p][s_local]  (x*dt*decay)
  __shared__ __align__(16) _Float16 sG[8][32 * 32]; // per-wave masked G

  size_t acsBase = (size_t)((e * NHEADS + h) * NCHUNK + c) * CHUNKSZ;
  sAcs[tid] = ACS[acsBase + tid];
  __syncthreads();
  float chsum = sAcs[CHUNKSZ - 1];

  // direction-aware global row of chunk-time tc
  int rowBase = (e < 2) ? (e * LSEQ + c * CHUNKSZ) : ((e - 2) * LSEQ + (LSEQ - 1) - c * CHUNKSZ);
  int sgn     = (e < 2) ? 1 : -1;
  int bOff = DINNER + ((e < 2) ? 0 : 128);
  int cOff = DINNER + ((e < 2) ? 64 : 192);

  // preload C A-frags for this wave's t rows (K = n = 64 -> 2 ksteps)
  v16h cfrag[2][2];
#pragma unroll
  for (int tt = 0; tt < 2; tt++) {
    int tc = wave * 32 + tt * 16 + col;
    const float* crow = XBC + (size_t)(rowBase + sgn * tc) * CONVDIM + cOff;
#pragma unroll
    for (int kk = 0; kk < 2; kk++) cfrag[tt][kk] = afrag_f32(crow + kk * 32, hh);
  }

  v8f yacc[2][4];
#pragma unroll
  for (int tt = 0; tt < 2; tt++)
#pragma unroll
    for (int pt = 0; pt < 4; pt++) yacc[tt][pt] = zero8();
  v8f stacc[2];
  stacc[0] = zero8(); stacc[1] = zero8();
  const int ptile = wave & 3;

  for (int s0 = 0; s0 < CHUNKSZ; s0 += 32) {
    __syncthreads();
    // stage B / X tiles (32 x 64)
    for (int j = tid; j < 2048; j += 256) {
      int sl = j >> 6, d = j & 63;
      int tc = s0 + sl;
      int grow = rowBase + sgn * tc;
      float dtv = DT[(size_t)(e * LSEQ + c * CHUNKSZ + tc) * NHEADS + h];
      float xv  = XBC[(size_t)grow * CONVDIM + h * HEADDIM + d] * dtv;
      float dec = __expf(chsum - sAcs[tc]);
      sXT[d * 32 + sl]  = (_Float16)xv;
      sXdT[d * 32 + sl] = (_Float16)(xv * dec);
      float bv = XBC[(size_t)grow * CONVDIM + bOff + d];
      sB[sl * 64 + d]  = (_Float16)bv;
      sBT[d * 32 + sl] = (_Float16)bv;
    }
    __syncthreads();

    // G = C . B^T for this wave's 32 t-rows, masked + exp'd, into wave-private LDS
#pragma unroll
    for (int tt = 0; tt < 2; tt++) {
#pragma unroll
      for (int st = 0; st < 2; st++) {
        v8f g = zero8();
#pragma unroll
        for (int kk = 0; kk < 2; kk++) {
          v16h bf = bfrag_h(&sB[(size_t)(st * 16 + col) * 64 + kk * 32], hh);
          g = WMMA_F16(cfrag[tt][kk], bf, g);
        }
#pragma unroll
        for (int r = 0; r < 8; r++) {
          int tloc = tt * 16 + r + 8 * hh;
          int tg = wave * 32 + tloc;
          int sg = s0 + st * 16 + col;
          float v = (sg <= tg) ? g[r] * __expf(sAcs[tg] - sAcs[sg]) : 0.0f;
          sG[wave][tloc * 32 + st * 16 + col] = (_Float16)v;
        }
      }
    }
    // wave-private LDS round trip; compiler inserts dscnt waits

    // Y_diag += (G*L) . X
#pragma unroll
    for (int tt = 0; tt < 2; tt++) {
      v16h af = afrag_h(&sG[wave][(size_t)(tt * 16 + col) * 32], hh);
#pragma unroll
      for (int pt = 0; pt < 4; pt++) {
        v16h bf = bfrag_h(&sXT[(size_t)(pt * 16 + col) * 32], hh);
        yacc[tt][pt] = WMMA_F16(af, bf, yacc[tt][pt]);
      }
    }
    // states[p,n] += Xd^T . B
    {
      v16h af = afrag_h(&sXdT[(size_t)(ptile * 16 + col) * 32], hh);
#pragma unroll
      for (int ii = 0; ii < 2; ii++) {
        int nt = (wave >> 2) * 2 + ii;
        v16h bf = bfrag_h(&sBT[(size_t)(nt * 16 + col) * 32], hh);
        stacc[ii] = WMMA_F16(af, bf, stacc[ii]);
      }
    }
  }

  // write Y_diag
#pragma unroll
  for (int tt = 0; tt < 2; tt++)
#pragma unroll
    for (int pt = 0; pt < 4; pt++)
#pragma unroll
      for (int r = 0; r < 8; r++) {
        int tg = wave * 32 + tt * 16 + r + 8 * hh;
        int p  = pt * 16 + col;
        YS[(size_t)(e * LSEQ + c * CHUNKSZ + tg) * DINNER + h * HEADDIM + p] = yacc[tt][pt][r];
      }
  // write chunk states
  size_t stBase = (size_t)((e * NHEADS + h) * NCHUNK + c) * (HEADDIM * DSTATE);
#pragma unroll
  for (int ii = 0; ii < 2; ii++)
#pragma unroll
    for (int r = 0; r < 8; r++) {
      int p = ptile * 16 + r + 8 * hh;
      int n = ((wave >> 2) * 2 + ii) * 16 + col;
      STATES[stBase + (size_t)p * DSTATE + n] = stacc[ii][r];
    }
}

// ---------------- 6) inter-chunk state recurrence ----------------
__global__ void __launch_bounds__(256) scan_states_kernel(const float* __restrict__ STATES,
                                                          const float* __restrict__ CHS,
                                                          float* __restrict__ SRUN) {
  int eh = blockIdx.x;  // e*32 + h
  int tid = threadIdx.x;
  float run[16];
#pragma unroll
  for (int k = 0; k < 16; k++) run[k] = 0.0f;
  for (int c = 0; c < NCHUNK; c++) {
    size_t base = ((size_t)eh * NCHUNK + c) * (HEADDIM * DSTATE);
    float dec = __expf(CHS[eh * NCHUNK + c]);
#pragma unroll
    for (int k = 0; k < 16; k++) {
      int j = tid + k * 256;
      SRUN[base + j] = run[k];                       // state entering chunk c
      run[k] = run[k] * dec + STATES[base + j];
    }
  }
}

// ---------------- 7) Y_off += exp(Acs[t]) * C . S_run^T (WMMA) ----------------
__global__ void __launch_bounds__(256) y_off_kernel(const float* __restrict__ XBC,
                                                    const float* __restrict__ ACS,
                                                    const float* __restrict__ SRUN,
                                                    float* __restrict__ YS) {
  int bid = blockIdx.x;
  int e = bid >> 8, h = (bid >> 3) & 31, c = bid & 7;
  int tid = threadIdx.x, wave = tid >> 5, lane = tid & 31;
  int col = lane & 15, hh = lane >> 4;

  __shared__ __align__(16) _Float16 sS[HEADDIM * DSTATE];  // [p][n]
  __shared__ float sAcs[CHUNKSZ];
  size_t acsBase = (size_t)((e * NHEADS + h) * NCHUNK + c) * CHUNKSZ;
  size_t srBase  = (size_t)((e * NHEADS + h) * NCHUNK + c) * (HEADDIM * DSTATE);
  sAcs[tid] = ACS[acsBase + tid];
  for (int j = tid; j < HEADDIM * DSTATE; j += 256) sS[j] = (_Float16)SRUN[srBase + j];
  __syncthreads();

  int rowBase = (e < 2) ? (e * LSEQ + c * CHUNKSZ) : ((e - 2) * LSEQ + (LSEQ - 1) - c * CHUNKSZ);
  int sgn     = (e < 2) ? 1 : -1;
  int cOff = DINNER + ((e < 2) ? 64 : 192);

  v8f acc[2][4];
#pragma unroll
  for (int tt = 0; tt < 2; tt++)
#pragma unroll
    for (int pt = 0; pt < 4; pt++) acc[tt][pt] = zero8();

#pragma unroll
  for (int tt = 0; tt < 2; tt++) {
    int tc = wave * 32 + tt * 16 + col;
    const float* crow = XBC + (size_t)(rowBase + sgn * tc) * CONVDIM + cOff;
#pragma unroll
    for (int kk = 0; kk < 2; kk++) {
      v16h af = afrag_f32(crow + kk * 32, hh);
#pragma unroll
      for (int pt = 0; pt < 4; pt++) {
        v16h bf = bfrag_h(&sS[(size_t)(pt * 16 + col) * DSTATE + kk * 32], hh);
        acc[tt][pt] = WMMA_F16(af, bf, acc[tt][pt]);
      }
    }
  }
#pragma unroll
  for (int tt = 0; tt < 2; tt++)
#pragma unroll
    for (int pt = 0; pt < 4; pt++)
#pragma unroll
      for (int r = 0; r < 8; r++) {
        int tg = wave * 32 + tt * 16 + r + 8 * hh;
        int p  = pt * 16 + col;
        float scale = __expf(sAcs[tg]);
        size_t idx = (size_t)(e * LSEQ + c * CHUNKSZ + tg) * DINNER + h * HEADDIM + p;
        YS[idx] += acc[tt][pt][r] * scale;
      }
}

// ---------------- 8) fc_D skip projection (LDS-staged, coalesced) ----------------
__global__ void __launch_bounds__(256) fcd_kernel(const float* __restrict__ XBC,
                                                  const float* __restrict__ fc_D_w,
                                                  const float* __restrict__ D_param,
                                                  float* __restrict__ FCD) {
  int row = blockIdx.x;  // b*2048 + l
  int tid = threadIdx.x;
  __shared__ float sx[DINNER];
  __shared__ float sred[256];
  for (int j = tid; j < DINNER; j += 256) sx[j] = XBC[(size_t)row * CONVDIM + j];
  __syncthreads();
  int hh = tid >> 3, part = tid & 7;         // 8 threads per head
  const float* w  = fc_D_w + (size_t)hh * DINNER + part * 256;
  const float* xs = sx + part * 256;
  float acc = 0.0f;
#pragma unroll 8
  for (int d = 0; d < 256; d++) acc += xs[d] * w[d];
  sred[tid] = acc;
  __syncthreads();
  if (part < 4) sred[tid] += sred[tid + 4];
  __syncthreads();
  if (part < 2) sred[tid] += sred[tid + 2];
  __syncthreads();
  if (part == 0)
    FCD[(size_t)row * NHEADS + hh] = sred[tid] + sred[tid + 1] + D_param[hh];
}

// ---------------- 9) roll + combine + RMSNorm + silu gate ----------------
__global__ void __launch_bounds__(256) combine_norm_kernel(const float* __restrict__ YS,
                                                           const float* __restrict__ XBC,
                                                           const float* __restrict__ FCD,
                                                           const float* __restrict__ ZX,
                                                           const float* __restrict__ norm_w,
                                                           float* __restrict__ YN) {
  int bid = blockIdx.x;  // b*2048 + l
  int b = bid >> 11, l = bid & (LSEQ - 1);
  int tid = threadIdx.x;
  float yc[8];
  float ss = 0.0f;
#pragma unroll
  for (int i = 0; i < 8; i++) {
    int d = tid + i * 256;
    float fw = (l > 0)        ? YS[(size_t)(b * LSEQ + l - 1) * DINNER + d] : 0.0f;
    float bw = (l < LSEQ - 1) ? YS[(size_t)((b + 2) * LSEQ + (LSEQ - 2 - l)) * DINNER + d] : 0.0f;
    float xo = XBC[(size_t)(b * LSEQ + l) * CONVDIM + d];
    float sk = xo * FCD[(size_t)(b * LSEQ + l) * NHEADS + (d >> 6)];
    yc[i] = fw + bw + sk;
    ss += yc[i] * yc[i];
  }
  __shared__ float red[256];
  red[tid] = ss;
  __syncthreads();
  for (int o = 128; o > 0; o >>= 1) {
    if (tid < o) red[tid] += red[tid + o];
    __syncthreads();
  }
  float rms = rsqrtf(red[0] / (float)DINNER + 1e-5f);
#pragma unroll
  for (int i = 0; i < 8; i++) {
    int d = tid + i * 256;
    float z = ZX[(size_t)(b * LSEQ + l) * DINPROJ + d];
    float sz = z / (1.0f + __expf(-z));
    YN[(size_t)(b * LSEQ + l) * DINNER + d] = yc[i] * rms * norm_w[d] * sz;
  }
}

// ---------------- launch ----------------
extern "C" void kernel_launch(void* const* d_in, const int* in_sizes, int n_in,
                              void* d_out, int out_size, void* d_ws, size_t ws_size,
                              hipStream_t stream) {
  (void)in_sizes; (void)n_in; (void)out_size; (void)ws_size;
  const float* u          = (const float*)d_in[0];
  const float* in_proj_w  = (const float*)d_in[1];
  const float* conv_w     = (const float*)d_in[2];
  const float* conv_b     = (const float*)d_in[3];
  const float* dt_bias    = (const float*)d_in[4];
  const float* A_log      = (const float*)d_in[5];
  const float* D_param    = (const float*)d_in[6];
  const float* fc_D_w     = (const float*)d_in[7];
  const float* norm_w     = (const float*)d_in[8];
  const float* out_proj_w = (const float*)d_in[9];

  float* ws  = (float*)d_ws;
  float* ZX  = ws + OFF_ZX;
  float* DTb = ws + OFF_DT;
  float* XBC = ws + OFF_XBC;
  float* ACS = ws + OFF_ACS;
  float* CHS = ws + OFF_CHS;
  float* YS  = ws + OFF_YS;
  float* ST  = ws + OFF_ST;
  float* SR  = ws + OFF_SR;
  float* FCD = ws + OFF_FCD;
  float* YN  = ws + OFF_YN;
  float* OUT = (float*)d_out;

  // 1) in_proj: [4096,1024] x [4416,1024]^T -> [4096,4416]
  wmma_gemm_xwT<<<dim3(DINPROJ / 64, MROWS / 128), 256, 0, stream>>>(u, in_proj_w, ZX,
                                                                     MROWS, DINPROJ, DMODEL);
  // 2) dt
  prep_dt_kernel<<<(NBD * LSEQ * NHEADS) / 256, 256, 0, stream>>>(ZX, dt_bias, DTb);
  // 3) conv + silu
  conv_silu_kernel<<<(BATCH * LSEQ * CONVDIM) / 256, 256, 0, stream>>>(ZX, conv_w, conv_b, XBC);
  // 4) chunk cumsum of dt*A
  chunk_cumsum_kernel<<<NBD * NHEADS * NCHUNK, 256, 0, stream>>>(DTb, A_log, ACS, CHS);
  // 5) intra-chunk SSD (Y_diag + states)
  ssd_chunk_kernel<<<NBD * NHEADS * NCHUNK, 256, 0, stream>>>(XBC, DTb, ACS, YS, ST);
  // 6) inter-chunk recurrence
  scan_states_kernel<<<NBD * NHEADS, 256, 0, stream>>>(ST, CHS, SR);
  // 7) Y_off
  y_off_kernel<<<NBD * NHEADS * NCHUNK, 256, 0, stream>>>(XBC, ACS, SR, YS);
  // 8) fc_D skip
  fcd_kernel<<<BATCH * LSEQ, 256, 0, stream>>>(XBC, fc_D_w, D_param, FCD);
  // 9) combine + RMSNorm + gate
  combine_norm_kernel<<<BATCH * LSEQ, 256, 0, stream>>>(YS, XBC, FCD, ZX, norm_w, YN);
  // 10) out_proj: [4096,2048] x [1024,2048]^T -> [4096,1024]
  wmma_gemm_xwT<<<dim3(DMODEL / 64, MROWS / 128), 256, 0, stream>>>(YN, out_proj_w, OUT,
                                                                    MROWS, DMODEL, DINNER);
}
